// PairWiseAttention_57698590654943
// MI455X (gfx1250) — matrix operand order repaired and checked by reading, not compiled, not run on hardware
//
#include <hip/hip_runtime.h>
#include <hip/hip_bf16.h>

// ---------------------------------------------------------------------------
// PairWiseAttention for MI455X (gfx1250), wave32 + V_WMMA_F32_16X16X32_BF16
// + Tensor Data Mover staging of the score GEMM's K operand into LDS.
//
//   q = in1 @ W1^T ; k = in2 @ W2^T            (bf16 WMMA, f32 accum)
//   S = q @ k^T    (f32, 128 MB -> L2-resident; K tiles via TDM->LDS)
//   row stats (max,sum) over N2 ; col stats over N1
//   out1 = softmax_row(S)  @ in2   (weights built in-register from S)
//   out2 = softmax_col(S)^T@ in1   (S^T tiles staged via LDS)
// ---------------------------------------------------------------------------

typedef __bf16 bf16_t;
typedef __attribute__((ext_vector_type(16))) __bf16        v16bf;
typedef __attribute__((ext_vector_type(8)))  float         v8f;
typedef __attribute__((ext_vector_type(4)))  unsigned int  v4u;
typedef __attribute__((ext_vector_type(8)))  int           v8i;
typedef __attribute__((ext_vector_type(4)))  int           v4i;

#define BATCH 8
#define NSEQ  2048
#define HID   512

// LDS row stride (elements) for the TDM-staged K tile: 32 data + 8 pad = 80 B
// (16B-aligned rows for ds_load_b128, 20-bank stride to spread conflicts).
#define KT_LDS_LD 40

__device__ __forceinline__ v8f wmma_bf16(v16bf a, v16bf b, v8f c) {
    return __builtin_amdgcn_wmma_f32_16x16x32_bf16(
        /*neg_a=*/false, a, /*neg_b=*/false, b,
        /*c_mod=*/(short)0, c, /*reuse_a=*/false, /*reuse_b=*/false);
}

// A-matrix 16x32 bf16 frag (M x K), row-major source, tile ptr = &A[m0][k0].
// Lane L: row M = L&15; VGPR0..3 = K=(L>>4)*8+0..7, VGPR4..7 = K=16+(L>>4)*8+0..7.
__device__ __forceinline__ v16bf load_a16x32(const bf16_t* __restrict__ tile, int ld) {
    int lane = threadIdx.x & 31;
    const bf16_t* r = tile + (size_t)(lane & 15) * ld + ((lane >> 4) << 3);
    union { v16bf v; uint4 q[2]; } u;
    u.q[0] = *(const uint4*)(r);
    u.q[1] = *(const uint4*)(r + 16);
    return u.v;
}

// B-matrix 32x16 bf16 frag (K x N) for X @ Y^T, Y row-major [N][K]:
// lane L: col = L&15, VGPR0..3 = K=(L>>4)*8+0..7, VGPR4..7 = K=16+(L>>4)*8+0..7.
__device__ __forceinline__ v16bf load_bT16x32(const bf16_t* __restrict__ ytile, int ld) {
    int lane = threadIdx.x & 31;
    const bf16_t* r = ytile + (size_t)(lane & 15) * ld + ((lane >> 4) << 3);
    union { v16bf v; uint4 q[2]; } u;
    u.q[0] = *(const uint4*)(r);
    u.q[1] = *(const uint4*)(r + 16);
    return u.v;
}

// C/D 16x16 f32: lane col = L&15, VGPR j -> row (L>>4)*8 + j.
__device__ __forceinline__ void store_d_f32(float* __restrict__ tile, int ld, v8f d) {
    int lane = threadIdx.x & 31;
    int col = lane & 15, r0 = (lane >> 4) << 3;
#pragma unroll
    for (int j = 0; j < 8; ++j) tile[(size_t)(r0 + j) * ld + col] = d[j];
}

__device__ __forceinline__ void store_d_bf16(bf16_t* __restrict__ tile, int ld, v8f d) {
    int lane = threadIdx.x & 31;
    int col = lane & 15, r0 = (lane >> 4) << 3;
#pragma unroll
    for (int j = 0; j < 8; ++j) tile[(size_t)(r0 + j) * ld + col] = (bf16_t)d[j];
}

// ---------------------------------------------------------------------------
// Tensor Data Mover: stage a 128-row x 32-col bf16 tile (row stride HID elems
// in memory) into LDS with 16B padding every 64B (-> 80B LDS row stride).
// This toolchain exposes the 6-arg builtin:
//   (uint32x4 g0, int32x8 g1, int32x4 g2, int32x4 g3, int32x8 g4, i32 cpol)
#if __has_builtin(__builtin_amdgcn_tensor_load_to_lds)
#define HAVE_TDM 1
__device__ __forceinline__ void tdm_load_ktile(unsigned lds_byte_off,
                                               const bf16_t* gptr) {
    unsigned long long ga = (unsigned long long)(uintptr_t)gptr;
    v4u g0;
    g0[0] = 1u;                                   // count=1, user descriptor
    g0[1] = lds_byte_off;                         // lds_addr
    g0[2] = (unsigned)(ga & 0xFFFFFFFFull);       // global_addr[31:0]
    g0[3] = (unsigned)((ga >> 32) & 0x01FFFFFFull) | 0x80000000u; // addr[56:32], type=2
    v8i g1;
    g1[0] = (int)((1u << 16)        // data_size = 2 bytes
                | (1u << 20)        // pad_enable
                | (3u << 22)        // pad_interval: every 64 B
                | (3u << 25));      // pad_amount: 4 DWORDs = 16 B
    g1[1] = (int)(HID  << 16);      // tensor_dim0 = 512 (contiguous dim)
    g1[2] = (int)(NSEQ << 16);      // tensor_dim1 = 2048 (rows)
    g1[3] = (int)(32   << 16);      // tile_dim0 = 32
    g1[4] = 128;                    // tile_dim1 = 128 rows
    g1[5] = HID;                    // tensor_dim0_stride = 512 elems
    g1[6] = 0;
    g1[7] = 0;
    v4i gz4 = {0, 0, 0, 0};         // 2D tensor: groups 2/3 unused
    v8i gz8 = {0, 0, 0, 0, 0, 0, 0, 0};
    __builtin_amdgcn_tensor_load_to_lds(g0, g1, gz4, gz4, gz8, 0);
}
#else
#define HAVE_TDM 0
#endif

// ---------------------------------------------------------------------------
// f32 -> bf16 flat convert (weights)
__global__ __launch_bounds__(256)
void k_cvt_bf16(const float* __restrict__ in, bf16_t* __restrict__ out, int n) {
    int i = blockIdx.x * 256 + threadIdx.x;
    int stride = gridDim.x * 256;
    for (; i < n; i += stride) out[i] = (bf16_t)in[i];
}

// [B][N][H] f32 -> [B][H][N] bf16, 32x32 LDS-tiled transpose.
__global__ __launch_bounds__(256)
void k_transpose_cvt(const float* __restrict__ in, bf16_t* __restrict__ out) {
    __shared__ float tile[32][33];
    int b  = blockIdx.z;
    int h0 = blockIdx.x * 32, n0 = blockIdx.y * 32;
    const float* src = in  + (size_t)b * NSEQ * HID;
    bf16_t*      dst = out + (size_t)b * NSEQ * HID;
    int tx = threadIdx.x & 31, ty = threadIdx.x >> 5;   // 32 x 8
#pragma unroll
    for (int i = 0; i < 32; i += 8)
        tile[ty + i][tx] = src[(size_t)(n0 + ty + i) * HID + h0 + tx];
    __syncthreads();
#pragma unroll
    for (int i = 0; i < 32; i += 8)
        dst[(size_t)(h0 + ty + i) * NSEQ + n0 + tx] = (bf16_t)tile[tx][ty + i];
}

// Y[r][o] = sum_h X[r][h] * W[o][h]; X f32 (cvt in-register), W bf16 [H][H].
__global__ __launch_bounds__(128)
void k_linear(const float* __restrict__ X, const bf16_t* __restrict__ W,
              bf16_t* __restrict__ Y) {
    int wave = threadIdx.x >> 5;
    int m0 = blockIdx.x * 64 + wave * 16;
    int n0 = blockIdx.y * 64;
    int lane = threadIdx.x & 31;
    v8f acc[4] = {};
    for (int k0 = 0; k0 < HID; k0 += 32) {
        const float* r = X + (size_t)(m0 + (lane & 15)) * HID + k0 + ((lane >> 4) << 3);
        union { v16bf v; bf16_t e[16]; } a;
#pragma unroll
        for (int i = 0; i < 8; ++i) a.e[i]     = (bf16_t)r[i];
#pragma unroll
        for (int i = 0; i < 8; ++i) a.e[8 + i] = (bf16_t)r[16 + i];
#pragma unroll
        for (int c = 0; c < 4; ++c) {
            v16bf bb = load_bT16x32(W + (size_t)(n0 + c * 16) * HID + k0, HID);
            acc[c] = wmma_bf16(a.v, bb, acc[c]);
        }
    }
#pragma unroll
    for (int c = 0; c < 4; ++c)
        store_d_bf16(Y + (size_t)m0 * HID + n0 + c * 16, HID, acc[c]);
}

// S[b] = Q[b] @ K[b]^T. Block = 4 waves, block tile 64(M) x 128(N).
// K operand tile (128 rows x 32 k) staged into LDS by the TDM, double-buffered
// on TENSORcnt; B-frags come from LDS (ds_load_b128), A-frags from global.
__global__ __launch_bounds__(128)
void k_score(const bf16_t* __restrict__ Q, const bf16_t* __restrict__ Km,
             float* __restrict__ S) {
    __shared__ bf16_t kt[2][128 * KT_LDS_LD];
    size_t bo = (size_t)blockIdx.z * NSEQ * HID;
    size_t bs = (size_t)blockIdx.z * NSEQ * NSEQ;
    int wave = threadIdx.x >> 5;
    int m0 = blockIdx.y * 64 + wave * 16;
    int n0 = blockIdx.x * 128;
    const bf16_t* gk = Km + bo + (size_t)n0 * HID;      // + k0 per step
    const bf16_t* ga = Q  + bo + (size_t)m0 * HID;
    bool issuer = (threadIdx.x < 32);

#if HAVE_TDM
    unsigned lds0 = (unsigned)(uintptr_t)&kt[0][0];
    unsigned lds1 = (unsigned)(uintptr_t)&kt[1][0];
    if (issuer) tdm_load_ktile(lds0, gk);
#endif
    v8f acc[8] = {};
    for (int ks = 0; ks < HID / 32; ++ks) {
#if HAVE_TDM
        if (issuer) __builtin_amdgcn_s_wait_tensorcnt(0);
        __syncthreads();                                 // tile ks visible
        if (issuer && (ks + 1) < HID / 32)               // overlap DMA ks+1
            tdm_load_ktile((ks & 1) ? lds0 : lds1, gk + (ks + 1) * 32);
#else
        __syncthreads();
        for (int i = threadIdx.x; i < 128 * 2; i += 128) {
            int row = i >> 1, seg = i & 1;
            *(uint4*)&kt[ks & 1][row * KT_LDS_LD + seg * 16] =
                *(const uint4*)(gk + (size_t)row * HID + ks * 32 + seg * 16);
        }
        __syncthreads();
#endif
        v16bf a = load_a16x32(ga + ks * 32, HID);
        const bf16_t* bufc = &kt[ks & 1][0];
#pragma unroll
        for (int c = 0; c < 8; ++c) {
            v16bf bb = load_bT16x32(bufc + (size_t)(c * 16) * KT_LDS_LD, KT_LDS_LD);
            acc[c] = wmma_bf16(a, bb, acc[c]);
        }
#if !HAVE_TDM
        __syncthreads();
#endif
    }
#pragma unroll
    for (int c = 0; c < 8; ++c)
        store_d_f32(S + bs + (size_t)m0 * NSEQ + n0 + c * 16, NSEQ, acc[c]);
}

// One block per global row (b*N+n): rowmax and rowsum of exp(s - max).
__global__ __launch_bounds__(256)
void k_row_stats(const float* __restrict__ S, float* __restrict__ rmax,
                 float* __restrict__ rsum) {
    const float* row = S + (size_t)blockIdx.x * NSEQ;
    __shared__ float red[256];
    int t = threadIdx.x;
    float m = -3.402823466e38f;
    for (int i = t; i < NSEQ; i += 256) m = fmaxf(m, row[i]);
    red[t] = m; __syncthreads();
    for (int s = 128; s > 0; s >>= 1) {
        if (t < s) red[t] = fmaxf(red[t], red[t + s]);
        __syncthreads();
    }
    m = red[0]; __syncthreads();
    float sum = 0.0f;
    for (int i = t; i < NSEQ; i += 256) sum += __expf(row[i] - m);
    red[t] = sum; __syncthreads();
    for (int s = 128; s > 0; s >>= 1) {
        if (t < s) red[t] += red[t + s];
        __syncthreads();
    }
    if (t == 0) { rmax[blockIdx.x] = m; rsum[blockIdx.x] = red[0]; }
}

// Column stats: thread owns a column; reads are lane-contiguous (coalesced).
__global__ __launch_bounds__(256)
void k_col_stats(const float* __restrict__ S, float* __restrict__ cmax,
                 float* __restrict__ csum) {
    int b = blockIdx.y;
    int c = blockIdx.x * 256 + threadIdx.x;
    const float* Sb = S + (size_t)b * NSEQ * NSEQ;
    float m = -3.402823466e38f;
    for (int n = 0; n < NSEQ; ++n) m = fmaxf(m, Sb[(size_t)n * NSEQ + c]);
    float sum = 0.0f;
    for (int n = 0; n < NSEQ; ++n) sum += __expf(Sb[(size_t)n * NSEQ + c] - m);
    cmax[b * NSEQ + c] = m;
    csum[b * NSEQ + c] = sum;
}

// out1[b] = softmax_row(S[b]) @ in2[b]. Wave tile 32(M) x 64(H): two A-frags
// (rows r0, r0+16) share each B-frag. A = bf16(exp(S - rmax) * 1/rsum).
__global__ __launch_bounds__(128)
void k_out1(const float* __restrict__ S, const float* __restrict__ rmax,
            const float* __restrict__ rsum, const bf16_t* __restrict__ V2T,
            float* __restrict__ O) {
    int b = blockIdx.z;
    const float*  Sb = S   + (size_t)b * NSEQ * NSEQ;
    const bf16_t* Vb = V2T + (size_t)b * NSEQ * HID;    // [H][N]
    float*        Ob = O   + (size_t)b * NSEQ * HID;
    int wave = threadIdx.x >> 5, lane = threadIdx.x & 31;
    int m0 = blockIdx.y * 128 + wave * 32;
    int h0 = blockIdx.x * 64;
    int r0 = m0 + (lane & 15), r1 = r0 + 16;
    int half = lane >> 4;
    float rm0 = rmax[b * NSEQ + r0], ri0 = 1.0f / rsum[b * NSEQ + r0];
    float rm1 = rmax[b * NSEQ + r1], ri1 = 1.0f / rsum[b * NSEQ + r1];
    v8f acc[2][4] = {};
    for (int k0 = 0; k0 < NSEQ; k0 += 32) {
        const float* s0 = Sb + (size_t)r0 * NSEQ + k0 + half * 8;
        const float* s1 = Sb + (size_t)r1 * NSEQ + k0 + half * 8;
        __builtin_prefetch(s0 + 64, 0, 0);               // global_prefetch_b8
        __builtin_prefetch(s1 + 64, 0, 0);
        union { v16bf v; bf16_t e[16]; } a0, a1;
#pragma unroll
        for (int i = 0; i < 8; ++i) {
            a0.e[i]     = (bf16_t)(__expf(s0[i]      - rm0) * ri0);
            a0.e[8 + i] = (bf16_t)(__expf(s0[16 + i] - rm0) * ri0);
            a1.e[i]     = (bf16_t)(__expf(s1[i]      - rm1) * ri1);
            a1.e[8 + i] = (bf16_t)(__expf(s1[16 + i] - rm1) * ri1);
        }
#pragma unroll
        for (int c = 0; c < 4; ++c) {
            v16bf bb = load_bT16x32(Vb + (size_t)(h0 + c * 16) * NSEQ + k0, NSEQ);
            acc[0][c] = wmma_bf16(a0.v, bb, acc[0][c]);
            acc[1][c] = wmma_bf16(a1.v, bb, acc[1][c]);
        }
    }
#pragma unroll
    for (int c = 0; c < 4; ++c) {
        store_d_f32(Ob + (size_t)m0 * HID + h0 + c * 16, HID, acc[0][c]);
        store_d_f32(Ob + (size_t)(m0 + 16) * HID + h0 + c * 16, HID, acc[1][c]);
    }
}

// out2[b][m] = sum_n exp(S[n][m]-cmax[m])/csum[m] * in1[b][n]. S^T tiles
// (32 k x 128 m) staged via LDS; wave tile 32(M) x 64(H).
__global__ __launch_bounds__(128)
void k_out2(const float* __restrict__ S, const float* __restrict__ cmax,
            const float* __restrict__ csum, const bf16_t* __restrict__ V1T,
            float* __restrict__ O) {
    __shared__ float st[32][132];
    int b = blockIdx.z;
    const float*  Sb = S   + (size_t)b * NSEQ * NSEQ;
    const bf16_t* Vb = V1T + (size_t)b * NSEQ * HID;    // [H][N]
    float*        Ob = O   + (size_t)b * NSEQ * HID;
    int wave = threadIdx.x >> 5, lane = threadIdx.x & 31;
    int mblk = blockIdx.y * 128;
    int m0 = mblk + wave * 32;
    int h0 = blockIdx.x * 64;
    int r0 = m0 + (lane & 15), r1 = r0 + 16;            // m indices (N2 dim)
    int w0 = wave * 32 + (lane & 15), w1 = w0 + 16;     // m offsets in block
    int half = lane >> 4;
    float cm0 = cmax[b * NSEQ + r0], ci0 = 1.0f / csum[b * NSEQ + r0];
    float cm1 = cmax[b * NSEQ + r1], ci1 = 1.0f / csum[b * NSEQ + r1];
    int tc = threadIdx.x;                               // cooperative col 0..127
    v8f acc[2][4] = {};
    for (int k0 = 0; k0 < NSEQ; k0 += 32) {
        __syncthreads();
#pragma unroll
        for (int r = 0; r < 32; ++r)
            st[r][tc] = Sb[(size_t)(k0 + r) * NSEQ + mblk + tc];
        __syncthreads();
        union { v16bf v; bf16_t e[16]; } a0, a1;
#pragma unroll
        for (int i = 0; i < 8; ++i) {
            a0.e[i]     = (bf16_t)(__expf(st[half * 8 + i][w0]      - cm0) * ci0);
            a0.e[8 + i] = (bf16_t)(__expf(st[16 + half * 8 + i][w0] - cm0) * ci0);
            a1.e[i]     = (bf16_t)(__expf(st[half * 8 + i][w1]      - cm1) * ci1);
            a1.e[8 + i] = (bf16_t)(__expf(st[16 + half * 8 + i][w1] - cm1) * ci1);
        }
#pragma unroll
        for (int c = 0; c < 4; ++c) {
            v16bf bb = load_bT16x32(Vb + (size_t)(h0 + c * 16) * NSEQ + k0, NSEQ);
            acc[0][c] = wmma_bf16(a0.v, bb, acc[0][c]);
            acc[1][c] = wmma_bf16(a1.v, bb, acc[1][c]);
        }
    }
#pragma unroll
    for (int c = 0; c < 4; ++c) {
        store_d_f32(Ob + (size_t)m0 * HID + h0 + c * 16, HID, acc[0][c]);
        store_d_f32(Ob + (size_t)(m0 + 16) * HID + h0 + c * 16, HID, acc[1][c]);
    }
}

// ---------------------------------------------------------------------------
extern "C" void kernel_launch(void* const* d_in, const int* in_sizes, int n_in,
                              void* d_out, int out_size, void* d_ws, size_t ws_size,
                              hipStream_t stream) {
    const float* in1 = (const float*)d_in[0];   // [B][N][H]
    const float* in2 = (const float*)d_in[1];   // [B][N][H]
    const float* W1  = (const float*)d_in[2];   // [H][H] (out,in)
    const float* W2  = (const float*)d_in[3];
    float* out = (float*)d_out;                 // out1 | out2, each B*N*H f32

    const size_t nBNH = (size_t)BATCH * NSEQ * HID;
    const size_t nHH  = (size_t)HID * HID;
    const size_t nS   = (size_t)BATCH * NSEQ * NSEQ;

    // Workspace carve-out (~193 MB total).
    char* ws = (char*)d_ws;
    auto carve = [&](size_t bytes) {
        char* p = ws;
        ws += (bytes + 255) & ~(size_t)255;
        return p;
    };
    bf16_t* W1bf = (bf16_t*)carve(nHH  * sizeof(bf16_t));
    bf16_t* W2bf = (bf16_t*)carve(nHH  * sizeof(bf16_t));
    bf16_t* in1T = (bf16_t*)carve(nBNH * sizeof(bf16_t));   // [B][H][N]
    bf16_t* in2T = (bf16_t*)carve(nBNH * sizeof(bf16_t));
    bf16_t* qbf  = (bf16_t*)carve(nBNH * sizeof(bf16_t));   // [B*N][H]
    bf16_t* kbf  = (bf16_t*)carve(nBNH * sizeof(bf16_t));
    float*  Smat = (float*)carve(nS * sizeof(float));       // 128 MB, L2-resident
    float*  rmax = (float*)carve((size_t)BATCH * NSEQ * sizeof(float));
    float*  rsum = (float*)carve((size_t)BATCH * NSEQ * sizeof(float));
    float*  cmaxv = (float*)carve((size_t)BATCH * NSEQ * sizeof(float));
    float*  csumv = (float*)carve((size_t)BATCH * NSEQ * sizeof(float));
    (void)ws_size; (void)in_sizes; (void)n_in; (void)out_size;

    // 1) weight convert + input transpose-convert
    k_cvt_bf16<<<256, 256, 0, stream>>>(W1, W1bf, (int)nHH);
    k_cvt_bf16<<<256, 256, 0, stream>>>(W2, W2bf, (int)nHH);
    dim3 tgrid(HID / 32, NSEQ / 32, BATCH);
    k_transpose_cvt<<<tgrid, 256, 0, stream>>>(in1, in1T);
    k_transpose_cvt<<<tgrid, 256, 0, stream>>>(in2, in2T);

    // 2) q, k linears (batch flattened: R = B*N rows)
    dim3 lgrid((BATCH * NSEQ) / 64, HID / 64);
    k_linear<<<lgrid, 128, 0, stream>>>(in1, W1bf, qbf);
    k_linear<<<lgrid, 128, 0, stream>>>(in2, W2bf, kbf);

    // 3) score S = q k^T (TDM-staged K tiles)
    dim3 sgrid(NSEQ / 128, NSEQ / 64, BATCH);
    k_score<<<sgrid, 128, 0, stream>>>(qbf, kbf, Smat);

    // 4) softmax stats (row over N2, col over N1)
    k_row_stats<<<BATCH * NSEQ, 256, 0, stream>>>(Smat, rmax, rsum);
    dim3 cgrid(NSEQ / 256, BATCH);
    k_col_stats<<<cgrid, 256, 0, stream>>>(Smat, cmaxv, csumv);

    // 5) fused softmax + output GEMMs
    dim3 ogrid(HID / 64, NSEQ / 128, BATCH);
    k_out1<<<ogrid, 128, 0, stream>>>(Smat, rmax, rsum, in2T, out);
    k_out2<<<ogrid, 128, 0, stream>>>(Smat, cmaxv, csumv, in1T, out + nBNH);
}
